// MultiHeadSelfAttention_20306605375868
// MI455X (gfx1250) — matrix operand order repaired and checked by reading, not compiled, not run on hardware
//
#include <hip/hip_runtime.h>
#include <cstdint>

// ---------------------------------------------------------------------------
// MHSA for gfx1250 (MI455X): f16 WMMA GEMMs + fused flash attention.
// B=4, L=2048, D_MODEL=1024, H=16, Dh=64. All dims divide tile sizes exactly.
// Data movement: async global->LDS (ASYNCcnt) for GEMM double buffering and
// TDM tensor_load_to_lds (TENSORcnt) for attention K tiles, when available.
// ---------------------------------------------------------------------------

typedef __attribute__((ext_vector_type(16))) _Float16 v16h;
typedef __attribute__((ext_vector_type(8)))  _Float16 v8h;
typedef __attribute__((ext_vector_type(4)))  _Float16 v4h;
typedef __attribute__((ext_vector_type(2)))  _Float16 v2h;
typedef __attribute__((ext_vector_type(8)))  float    v8f;
typedef __attribute__((ext_vector_type(4)))  unsigned int v4u;
typedef __attribute__((ext_vector_type(8)))  int v8i;
typedef __attribute__((ext_vector_type(4)))  int v4i;

// address-space-qualified v4i for the async global->LDS b128 builtin
typedef __attribute__((address_space(1))) v4i gv4i;
typedef __attribute__((address_space(3))) v4i lv4i;

#define D_MODEL   1024
#define SEQ_L     2048
#define NBATCH    4
#define NHEADS    16
#define HEAD_DIM  64
#define MROWS     (NBATCH * SEQ_L)      // 8192

#ifndef __has_builtin
#define __has_builtin(x) 0
#endif
#if __has_builtin(__builtin_amdgcn_tensor_load_to_lds) && \
    __has_builtin(__builtin_amdgcn_s_wait_tensorcnt)
#define HAVE_TDM 1
#else
#define HAVE_TDM 0
#endif
#if __has_builtin(__builtin_amdgcn_global_load_async_to_lds_b128)
#define HAVE_ASYNC 1
#else
#define HAVE_ASYNC 0
#endif

// Generic -> AS(1)/AS(3) typed pointer conversion. AMDGPU generic LDS
// pointers hold the LDS byte offset in their low 32 bits (ISA 10.2), so
// integer round-trips are exact for both spaces.
#define TO_G128(p) ((gv4i*)(uintptr_t)(p))
#define TO_L128(p) ((lv4i*)(uint32_t)(uintptr_t)(p))

#if HAVE_ASYNC
__device__ __forceinline__ void wait_asynccnt0() {
#if __has_builtin(__builtin_amdgcn_s_wait_asynccnt)
  __builtin_amdgcn_s_wait_asynccnt(0);
#else
  asm volatile("s_wait_asynccnt 0" ::: "memory");
#endif
}
#endif

#if HAVE_TDM
// 2D TDM load: tile 64(d) x 64(rows) of f16, global row stride D_MODEL elems,
// into LDS with 16B padding after each 128B row (-> 72-half LDS row stride).
__device__ __forceinline__ void tdm_load_k_tile(uint32_t lds_off, const _Float16* gptr) {
  uint64_t ga = (uint64_t)(uintptr_t)gptr;
  v4u g0 = { 1u,                                  // count=1 (valid descriptor)
             lds_off,                             // lds_addr
             (uint32_t)ga,                        // global_addr[31:0]
             (uint32_t)(ga >> 32) | 0x80000000u };// global_addr[56:32] | type=2
  v8i g1 = { (1 << 16) | (1 << 20) | (4 << 22) | (3 << 25),
             //          ^data_size=2B  ^pad_en  ^pad every 32 dw  ^pad 4 dw
             (int)(64u << 16),     // tensor_dim0 = 64 (low half)
             (int)(64u << 16),     // tensor_dim1 = 64 (low half)
             (int)(64u << 16),     // tile_dim0 = 64
             64,                   // tile_dim1 = 64, tile_dim2 = 0
             D_MODEL,              // tensor_dim0_stride = 1024 elems
             0, 0 };
  v4i z = { 0, 0, 0, 0 };
#if defined(__clang_major__) && __clang_major__ >= 23
  v8i z8 = { 0, 0, 0, 0, 0, 0, 0, 0 };
  __builtin_amdgcn_tensor_load_to_lds(g0, g1, z, z, z8, 0);
#else
  __builtin_amdgcn_tensor_load_to_lds(g0, g1, z, z, 0);
#endif
}
#endif

// ---- fragment loaders (CDNA5 WMMA 16x16x32 f16 VGPR layouts) --------------
__device__ __forceinline__ v16h load_a_frag(const _Float16* p0, int stride, int lane) {
  const _Float16* p = p0 + (lane & 15) * stride + ((lane >> 4) << 3);
  v8h lo = *(const v8h*)(p);
  v8h hi = *(const v8h*)(p + 16);
  return __builtin_shufflevector(lo, hi, 0,1,2,3,4,5,6,7,8,9,10,11,12,13,14,15);
}

__device__ __forceinline__ v16h load_b_frag(const _Float16* p0, int stride, int lane) {
  const _Float16* p = p0 + (lane & 15) * stride + ((lane >> 4) << 3);
  v8h lo = *(const v8h*)(p);
  v8h hi = *(const v8h*)(p + 16);
  return __builtin_shufflevector(lo, hi, 0,1,2,3,4,5,6,7,8,9,10,11,12,13,14,15);
}

// ---- fp32 -> f16 bulk convert ---------------------------------------------
__global__ __launch_bounds__(256) void convert_f32_f16(const float* __restrict__ in,
                                                       _Float16* __restrict__ out) {
  int i = (blockIdx.x * 256 + threadIdx.x) * 4;
  float4 f = *(const float4*)(in + i);
  v4h o = { (_Float16)f.x, (_Float16)f.y, (_Float16)f.z, (_Float16)f.w };
  *(v4h*)(out + i) = o;
}

// ---- transpose + convert: W[k][n] f32 -> Wt[n][k] f16 ---------------------
__global__ __launch_bounds__(256) void transpose_conv(const float* __restrict__ W,
                                                      _Float16* __restrict__ Wt) {
  __shared__ __align__(16) _Float16 t[32][33];
  int n0 = blockIdx.x * 32, k0 = blockIdx.y * 32;
  int tx = threadIdx.x, ty = threadIdx.y;       // block (32, 8)
  #pragma unroll
  for (int i = 0; i < 4; ++i) {
    int k = ty + i * 8;
    t[k][tx] = (_Float16)W[(size_t)(k0 + k) * D_MODEL + n0 + tx];
  }
  __syncthreads();
  #pragma unroll
  for (int i = 0; i < 4; ++i) {
    int n = ty + i * 8;
    Wt[(size_t)(n0 + n) * D_MODEL + k0 + tx] = t[tx][n];
  }
}

// ---- 128x128-tile GEMM: C[M,1024] = A[M,1024](f16) * Bt(col-major f16) ----
#define LDS_STRIDE 48   // 32 halfs + 16 pad: 16B aligned rows, spreads banks
#if HAVE_ASYNC
#define NBUF 2
#else
#define NBUF 1
#endif

template <bool F16OUT>
__global__ __launch_bounds__(256) void gemm128(const _Float16* __restrict__ A,
                                               const _Float16* __restrict__ Bt,
                                               const float* __restrict__ bias,
                                               void* __restrict__ Cout, float scale) {
  __shared__ __align__(16) _Float16 sA[NBUF][128 * LDS_STRIDE];
  __shared__ __align__(16) _Float16 sB[NBUF][128 * LDS_STRIDE];
  const int tid  = threadIdx.x;
  const int lane = tid & 31, wave = tid >> 5;
  const int wm   = wave >> 2, wn = wave & 3;
  const int m0   = blockIdx.y * 128, n0 = blockIdx.x * 128;

  v8f acc[4][2] = {};

  const int srow = tid >> 1;              // 0..127
  const int sseg = (tid & 1) * 16;        // 0 / 16
  const int soff = srow * LDS_STRIDE + sseg;
  const _Float16* gA = A  + (size_t)(m0 + srow) * D_MODEL + sseg;
  const _Float16* gB = Bt + (size_t)(n0 + srow) * D_MODEL + sseg;

  auto compute = [&](int b) {
    v16h bf[2];
    #pragma unroll
    for (int j = 0; j < 2; ++j)
      bf[j] = load_b_frag(&sB[b][(wn * 32 + j * 16) * LDS_STRIDE], LDS_STRIDE, lane);
    #pragma unroll
    for (int i = 0; i < 4; ++i) {
      v16h af = load_a_frag(&sA[b][(wm * 64 + i * 16) * LDS_STRIDE], LDS_STRIDE, lane);
      #pragma unroll
      for (int j = 0; j < 2; ++j)
        acc[i][j] = __builtin_amdgcn_wmma_f32_16x16x32_f16(
            false, af, false, bf[j], (short)0, acc[i][j], false, false);
    }
  };

#if HAVE_ASYNC
  auto stage = [&](int k0, int b) {
    __builtin_amdgcn_global_load_async_to_lds_b128(TO_G128(gA + k0),     TO_L128(&sA[b][soff]),     0, 0);
    __builtin_amdgcn_global_load_async_to_lds_b128(TO_G128(gA + k0 + 8), TO_L128(&sA[b][soff + 8]), 0, 0);
    __builtin_amdgcn_global_load_async_to_lds_b128(TO_G128(gB + k0),     TO_L128(&sB[b][soff]),     0, 0);
    __builtin_amdgcn_global_load_async_to_lds_b128(TO_G128(gB + k0 + 8), TO_L128(&sB[b][soff + 8]), 0, 0);
  };
  // double-buffered: prefetch k0+32 while WMMAs run on k0
  stage(0, 0);
  wait_asynccnt0();
  __syncthreads();
  int buf = 0;
  for (int k0 = 0; k0 < D_MODEL; k0 += 32) {
    if (k0 + 32 < D_MODEL) stage(k0 + 32, buf ^ 1);
    compute(buf);
    wait_asynccnt0();
    __syncthreads();
    buf ^= 1;
  }
#else
  for (int k0 = 0; k0 < D_MODEL; k0 += 32) {
    __syncthreads();
    *(v8h*)&sA[0][soff]     = *(const v8h*)(gA + k0);
    *(v8h*)&sA[0][soff + 8] = *(const v8h*)(gA + k0 + 8);
    *(v8h*)&sB[0][soff]     = *(const v8h*)(gB + k0);
    *(v8h*)&sB[0][soff + 8] = *(const v8h*)(gB + k0 + 8);
    __syncthreads();
    compute(0);
  }
#endif

  const int lo = lane & 15, hi = lane >> 4;
  #pragma unroll
  for (int i = 0; i < 4; ++i)
    #pragma unroll
    for (int j = 0; j < 2; ++j) {
      int col = n0 + wn * 32 + j * 16 + lo;
      float bb = bias[col];
      #pragma unroll
      for (int v = 0; v < 8; ++v) {
        int row = m0 + wm * 64 + i * 16 + hi * 8 + v;
        float val = (acc[i][j][v] + bb) * scale;
        if (F16OUT) ((_Float16*)Cout)[(size_t)row * D_MODEL + col] = (_Float16)val;
        else        ((float*)Cout)   [(size_t)row * D_MODEL + col] = val;
      }
    }
}

// ---- fused flash attention -------------------------------------------------
#define KSTR 72   // 64 halfs + 8 pad (144 B rows, 16B aligned)

__global__ __launch_bounds__(256) void attn_flash(const _Float16* __restrict__ Qh,
                                                  const _Float16* __restrict__ Kh,
                                                  const _Float16* __restrict__ Vh,
                                                  _Float16* __restrict__ Ctx) {
  __shared__ __align__(16) _Float16 sK[64 * KSTR];        // [key][d]
  __shared__ __align__(16) _Float16 sV[64 * KSTR];        // [d][key] (transposed)
  __shared__ __align__(16) _Float16 sP[8 * 16 * KSTR];    // per-wave P tiles

  const int tid  = threadIdx.x;
  const int lane = tid & 31, wave = tid >> 5;
  const int lo = lane & 15, hi = lane >> 4;
  const int b = blockIdx.y >> 4, h = blockIdx.y & 15;
  const int q0 = blockIdx.x * 128 + wave * 16;
  const size_t base = (size_t)b * SEQ_L * D_MODEL + (size_t)h * HEAD_DIM;

  const _Float16* qp = Qh + base + (size_t)q0 * D_MODEL;
  v16h qf0 = load_a_frag(qp,      D_MODEL, lane);
  v16h qf1 = load_a_frag(qp + 32, D_MODEL, lane);

  float m[8], l[8];
  #pragma unroll
  for (int v = 0; v < 8; ++v) { m[v] = -1e30f; l[v] = 0.f; }
  v8f oacc[4] = {};

  _Float16* sPw = &sP[wave * 16 * KSTR];

#if !HAVE_TDM
  const int kr = tid >> 2, ks = (tid & 3) * 16;     // manual K staging indices
#endif
  const int vp  = (tid & 31) * 2;                    // V staging: key pair
  const int vds = (tid >> 5) * 8;                    // V staging: d segment

  for (int kb0 = 0; kb0 < SEQ_L; kb0 += 64) {
    __syncthreads();
#if HAVE_TDM
    if (wave == 0) {
      tdm_load_k_tile((uint32_t)(uintptr_t)&sK[0], Kh + base + (size_t)kb0 * D_MODEL);
      __builtin_amdgcn_s_wait_tensorcnt(0);
    }
#else
    const _Float16* gk = Kh + base + (size_t)(kb0 + kr) * D_MODEL + ks;
    *(v8h*)&sK[kr * KSTR + ks]     = *(const v8h*)(gk);
    *(v8h*)&sK[kr * KSTR + ks + 8] = *(const v8h*)(gk + 8);
#endif
    // V tile transposed: each thread packs 2 adjacent keys per d -> b32 stores
    const _Float16* gv = Vh + base + (size_t)(kb0 + vp) * D_MODEL + vds;
    v8h va = *(const v8h*)(gv);
    v8h vb = *(const v8h*)(gv + D_MODEL);
    #pragma unroll
    for (int j = 0; j < 8; ++j)
      *(v2h*)&sV[(vds + j) * KSTR + vp] = (v2h){ va[j], vb[j] };
    __syncthreads();

    // scores: S = Q * K^T
    v8f sacc[4];
    #pragma unroll
    for (int nt = 0; nt < 4; ++nt) {
      v16h kf0 = load_b_frag(&sK[(nt * 16) * KSTR],      KSTR, lane);
      v16h kf1 = load_b_frag(&sK[(nt * 16) * KSTR + 32], KSTR, lane);
      v8f z = {};
      z = __builtin_amdgcn_wmma_f32_16x16x32_f16(false, qf0, false, kf0, (short)0, z, false, false);
      z = __builtin_amdgcn_wmma_f32_16x16x32_f16(false, qf1, false, kf1, (short)0, z, false, false);
      sacc[nt] = z;
    }

    // online softmax
    float rmax[8];
    #pragma unroll
    for (int v = 0; v < 8; ++v) {
      rmax[v] = fmaxf(fmaxf(sacc[0][v], sacc[1][v]), fmaxf(sacc[2][v], sacc[3][v]));
      #pragma unroll
      for (int off = 1; off < 16; off <<= 1)
        rmax[v] = fmaxf(rmax[v], __shfl_xor(rmax[v], off, 32));
    }
    float alpha[8];
    #pragma unroll
    for (int v = 0; v < 8; ++v) {
      float mn = fmaxf(m[v], rmax[v]);
      alpha[v] = __expf(m[v] - mn);
      m[v] = mn;
    }
    float rsum[8] = {};
    #pragma unroll
    for (int nt = 0; nt < 4; ++nt)
      #pragma unroll
      for (int v = 0; v < 8; ++v) {
        float e = __expf(sacc[nt][v] - m[v]);
        rsum[v] += e;
        sPw[(hi * 8 + v) * KSTR + nt * 16 + lo] = (_Float16)e;
      }
    #pragma unroll
    for (int v = 0; v < 8; ++v) {
      #pragma unroll
      for (int off = 1; off < 16; off <<= 1)
        rsum[v] += __shfl_xor(rsum[v], off, 32);
      l[v] = l[v] * alpha[v] + rsum[v];
    }
    #pragma unroll
    for (int nt = 0; nt < 4; ++nt)
      #pragma unroll
      for (int v = 0; v < 8; ++v)
        oacc[nt][v] *= alpha[v];

    // O += P * V
    v16h pf0 = load_a_frag(sPw,      KSTR, lane);
    v16h pf1 = load_a_frag(sPw + 32, KSTR, lane);
    #pragma unroll
    for (int nt = 0; nt < 4; ++nt) {
      v16h vf0 = load_b_frag(&sV[(nt * 16) * KSTR],      KSTR, lane);
      v16h vf1 = load_b_frag(&sV[(nt * 16) * KSTR + 32], KSTR, lane);
      oacc[nt] = __builtin_amdgcn_wmma_f32_16x16x32_f16(false, pf0, false, vf0, (short)0, oacc[nt], false, false);
      oacc[nt] = __builtin_amdgcn_wmma_f32_16x16x32_f16(false, pf1, false, vf1, (short)0, oacc[nt], false, false);
    }
  }

  #pragma unroll
  for (int v = 0; v < 8; ++v) l[v] = 1.f / l[v];
  #pragma unroll
  for (int nt = 0; nt < 4; ++nt)
    #pragma unroll
    for (int v = 0; v < 8; ++v) {
      int row = q0 + hi * 8 + v;
      int col = nt * 16 + lo;
      Ctx[base + (size_t)row * D_MODEL + col] = (_Float16)(oacc[nt][v] * l[v]);
    }
}

// ---------------------------------------------------------------------------
extern "C" void kernel_launch(void* const* d_in, const int* in_sizes, int n_in,
                              void* d_out, int out_size, void* d_ws, size_t ws_size,
                              hipStream_t stream) {
  (void)in_sizes; (void)n_in; (void)out_size; (void)ws_size;
  const float* x  = (const float*)d_in[0];
  const float* Wq = (const float*)d_in[1];
  const float* bq = (const float*)d_in[2];
  const float* Wk = (const float*)d_in[3];
  const float* bk = (const float*)d_in[4];
  const float* Wv = (const float*)d_in[5];
  const float* bv = (const float*)d_in[6];
  const float* Wo = (const float*)d_in[7];
  const float* bo = (const float*)d_in[8];
  float* out = (float*)d_out;

  _Float16* Xh  = (_Float16*)d_ws;
  _Float16* Wqt = Xh  + (size_t)MROWS * D_MODEL;
  _Float16* Wkt = Wqt + (size_t)D_MODEL * D_MODEL;
  _Float16* Wvt = Wkt + (size_t)D_MODEL * D_MODEL;
  _Float16* Wot = Wvt + (size_t)D_MODEL * D_MODEL;
  _Float16* Qh  = Wot + (size_t)D_MODEL * D_MODEL;
  _Float16* Kh  = Qh  + (size_t)MROWS * D_MODEL;
  _Float16* Vh  = Kh  + (size_t)MROWS * D_MODEL;
  _Float16* Ctx = Vh  + (size_t)MROWS * D_MODEL;

  convert_f32_f16<<<(MROWS * D_MODEL) / (256 * 4), 256, 0, stream>>>(x, Xh);
  dim3 tb(32, 8), tg(D_MODEL / 32, D_MODEL / 32);
  transpose_conv<<<tg, tb, 0, stream>>>(Wq, Wqt);
  transpose_conv<<<tg, tb, 0, stream>>>(Wk, Wkt);
  transpose_conv<<<tg, tb, 0, stream>>>(Wv, Wvt);
  transpose_conv<<<tg, tb, 0, stream>>>(Wo, Wot);

  dim3 gg(D_MODEL / 128, MROWS / 128);
  gemm128<true><<<gg, 256, 0, stream>>>(Xh, Wqt, bq, Qh, 0.125f);
  gemm128<true><<<gg, 256, 0, stream>>>(Xh, Wkt, bk, Kh, 1.0f);
  gemm128<true><<<gg, 256, 0, stream>>>(Xh, Wvt, bv, Vh, 1.0f);

  attn_flash<<<dim3(SEQ_L / 128, NBATCH * NHEADS), 256, 0, stream>>>(Qh, Kh, Vh, Ctx);

  gemm128<false><<<gg, 256, 0, stream>>>(Ctx, Wot, bo, out, 1.0f);
}
